// AttentionalPooling_38929583571149
// MI455X (gfx1250) — compile-verified, matrix operand order checked by reading, not run on hardware
//
#include <hip/hip_runtime.h>
#include <hip/hip_bf16.h>
#include <stdint.h>

// Problem constants from the reference
#define BB 4
#define LL 1024
#define DD 256
#define UU 32

typedef __attribute__((ext_vector_type(16))) __bf16 v16bf;
typedef __attribute__((ext_vector_type(8)))  float  v8f;

union V16BF {
    uint32_t u[8];
    v16bf    v;
};

#define TWO_LOG2E 2.885390081777926815f   // 2*log2(e): tanh(x) needs exp2(2x*log2e)

// ---------------------------------------------------------------------------
// Kernel 1: q = X @ Wt, k = X @ Wx   (one wave per row, lane = u)
// ---------------------------------------------------------------------------
__global__ void qk_kernel(const float* __restrict__ x,
                          const float* __restrict__ Wt,
                          const float* __restrict__ Wx,
                          float* __restrict__ q,
                          float* __restrict__ k) {
    int row  = blockIdx.x * (blockDim.x >> 5) + (threadIdx.x >> 5); // 0..B*L-1
    int u    = threadIdx.x & 31;
    const float* xr = x + (size_t)row * DD;
    float qa = 0.0f, ka = 0.0f;
    for (int d = 0; d < DD; ++d) {
        float xv = xr[d];
        qa = fmaf(xv, Wt[d * UU + u], qa);
        ka = fmaf(xv, Wx[d * UU + u], ka);
    }
    q[(size_t)row * UU + u] = qa;
    k[(size_t)row * UU + u] = ka;
}

// ---------------------------------------------------------------------------
// Kernel 2: Xt_bf[b][d][l] = (bf16) X[b][l][d]   (K-contiguous WMMA B operand)
// ---------------------------------------------------------------------------
__global__ void xt_kernel(const float* __restrict__ x, __bf16* __restrict__ xt) {
    size_t idx = (size_t)blockIdx.x * blockDim.x + threadIdx.x;   // over B*L*D
    int d = (int)(idx & (DD - 1));
    size_t bl = idx >> 8;                     // b*L + l
    int l = (int)(bl & (LL - 1));
    int b = (int)(bl >> 10);
    xt[((size_t)b * DD + d) * LL + l] = (__bf16)x[idx];
}

// ---------------------------------------------------------------------------
// Kernel 3: per output row (b,i):
//   e_j = ba + Sum_u Wa_u * tanh(q_iu + k_ju + bh_u)
// Using tanh(y) = 1 - 2/(exp2(2*log2e*y)+1):
//   e_j = (ba + SumWa) + Sum_u (-2*Wa_u) * rcp(exp2(fma(k, C, qb2_u)) + 1)
// where qb2_u = (q_iu + bh_u)*C.  Per element: 3 VALU + 2 TRANS.
// Then stable softmax over j, store P (bf16).
// ---------------------------------------------------------------------------
__global__ void score_softmax_kernel(const float* __restrict__ q,
                                     const float* __restrict__ k,
                                     const float* __restrict__ bh,
                                     const float* __restrict__ Wa,
                                     const float* __restrict__ ba,
                                     __bf16* __restrict__ P) {
    int bi  = blockIdx.x;            // b*L + i
    int tid = threadIdx.x;           // 0..255
    __shared__ float qb2[UU], wa2[UU];
    __shared__ float ebias_s;
    __shared__ float red[256];

    if (tid < UU) {
        qb2[tid] = (q[(size_t)bi * UU + tid] + bh[tid]) * TWO_LOG2E;
        wa2[tid] = -2.0f * Wa[tid];
    }
    if (tid == 0) {
        float ws = 0.0f;
        for (int u = 0; u < UU; ++u) ws += Wa[u];
        ebias_s = ba[0] + ws;        // constant part of every score
    }
    __syncthreads();

    int b = bi >> 10;
    const float* kbase = k + (size_t)b * LL * UU;
    float ebias = ebias_s;

    float e[4];
    float lmax = -3.402823466e38f;
    #pragma unroll
    for (int r = 0; r < 4; ++r) {
        int j = tid + r * 256;
        const float4* kr = (const float4*)(kbase + (size_t)j * UU);
        float acc = 0.0f;
        #pragma unroll
        for (int u4 = 0; u4 < 8; ++u4) {
            float4 kv = kr[u4];
            int u = u4 * 4;
            {
                float t = __builtin_amdgcn_exp2f(fmaf(kv.x, TWO_LOG2E, qb2[u + 0]));
                acc = fmaf(wa2[u + 0], __builtin_amdgcn_rcpf(t + 1.0f), acc);
            }
            {
                float t = __builtin_amdgcn_exp2f(fmaf(kv.y, TWO_LOG2E, qb2[u + 1]));
                acc = fmaf(wa2[u + 1], __builtin_amdgcn_rcpf(t + 1.0f), acc);
            }
            {
                float t = __builtin_amdgcn_exp2f(fmaf(kv.z, TWO_LOG2E, qb2[u + 2]));
                acc = fmaf(wa2[u + 2], __builtin_amdgcn_rcpf(t + 1.0f), acc);
            }
            {
                float t = __builtin_amdgcn_exp2f(fmaf(kv.w, TWO_LOG2E, qb2[u + 3]));
                acc = fmaf(wa2[u + 3], __builtin_amdgcn_rcpf(t + 1.0f), acc);
            }
        }
        e[r] = acc + ebias;
        lmax = fmaxf(lmax, e[r]);
    }

    // block max
    red[tid] = lmax; __syncthreads();
    for (int s = 128; s > 0; s >>= 1) {
        if (tid < s) red[tid] = fmaxf(red[tid], red[tid + s]);
        __syncthreads();
    }
    float m = red[0];
    __syncthreads();

    float p[4];
    float lsum = 0.0f;
    #pragma unroll
    for (int r = 0; r < 4; ++r) { p[r] = __expf(e[r] - m); lsum += p[r]; }

    // block sum
    red[tid] = lsum; __syncthreads();
    for (int s = 128; s > 0; s >>= 1) {
        if (tid < s) red[tid] += red[tid + s];
        __syncthreads();
    }
    float inv = __fdividef(1.0f, red[0]);

    __bf16* Pr = P + (size_t)bi * LL;
    #pragma unroll
    for (int r = 0; r < 4; ++r)
        Pr[tid + r * 256] = (__bf16)(p[r] * inv);
}

// ---------------------------------------------------------------------------
// Kernel 4: V = P @ X via v_wmma_f32_16x16x32_bf16.
// One wave per 16x64 output tile (4 accumulators share one A load per k-step):
// 10 vmem dwords per WMMA instead of 16.  K loop over L=1024, step 32.
// 16-bit operand layout (ISA 7.12.2): lane m = lane&15, kbase = (lane>=16)?8:0,
// VGPR v<4 holds K = kbase+2v,2v+1 ; v>=4 holds K = 16+kbase+2(v-4),+1.
// ---------------------------------------------------------------------------
__global__ void pv_wmma_kernel(const __bf16* __restrict__ P,
                               const __bf16* __restrict__ Xt,
                               float* __restrict__ out) {
    int wid  = blockIdx.x * (blockDim.x >> 5) + (threadIdx.x >> 5); // 0..1023
    int dg   = wid & 3;           // 4 groups of 64 columns over D=256
    int it   = (wid >> 2) & 63;   // 64 i-tiles of 16 over L=1024
    int b    = wid >> 8;          // batch
    int lane = threadIdx.x & 31;

    int mr = lane & 15;
    int kb = (lane >> 4) << 3;    // 0 or 8

    const __bf16* Pa  = P  + ((size_t)b * LL + it * 16 + mr) * LL;      // A row
    const __bf16* Xb0 = Xt + ((size_t)b * DD + dg * 64 + mr) * LL;      // B n-tile 0

    v8f acc0 = {}, acc1 = {}, acc2 = {}, acc3 = {};
    for (int k0 = 0; k0 < LL; k0 += 32) {
        V16BF a, b0, b1, b2, b3;
        #pragma unroll
        for (int v = 0; v < 8; ++v) {
            int kk = (v < 4) ? (kb + 2 * v) : (16 + kb + 2 * (v - 4));
            a.u[v]  = *(const uint32_t*)(Pa  + k0 + kk);
            b0.u[v] = *(const uint32_t*)(Xb0 +  0 * LL + k0 + kk);
            b1.u[v] = *(const uint32_t*)(Xb0 + 16 * LL + k0 + kk);
            b2.u[v] = *(const uint32_t*)(Xb0 + 32 * LL + k0 + kk);
            b3.u[v] = *(const uint32_t*)(Xb0 + 48 * LL + k0 + kk);
        }
        acc0 = __builtin_amdgcn_wmma_f32_16x16x32_bf16(false, a.v, false, b0.v, (short)0, acc0, false, false);
        acc1 = __builtin_amdgcn_wmma_f32_16x16x32_bf16(false, a.v, false, b1.v, (short)0, acc1, false, false);
        acc2 = __builtin_amdgcn_wmma_f32_16x16x32_bf16(false, a.v, false, b2.v, (short)0, acc2, false, false);
        acc3 = __builtin_amdgcn_wmma_f32_16x16x32_bf16(false, a.v, false, b3.v, (short)0, acc3, false, false);
    }

    // C/D layout: VGPR r -> M = r + 8*(lane>=16), N = lane&15
    int nh = lane >> 4;
    int n  = lane & 15;
    #pragma unroll
    for (int r = 0; r < 8; ++r) {
        int m = r + 8 * nh;
        float* o = out + ((size_t)b * LL + it * 16 + m) * DD + dg * 64 + n;
        o[0]  = acc0[r];
        o[16] = acc1[r];
        o[32] = acc2[r];
        o[48] = acc3[r];
    }
}

// ---------------------------------------------------------------------------
// Launch. Inputs (fp32): 0=inputs[B,L,D] 1=Wt[D,U] 2=Wx[D,U] 3=bh[U]
//                        4=Wa[U,1] 5=ba[1].  Output: v[B,L,D] fp32.
// Workspace (bytes): q @0 (512K), k @512K (512K), Xt_bf @1M (2M), P_bf @3M (8M)
// ---------------------------------------------------------------------------
extern "C" void kernel_launch(void* const* d_in, const int* in_sizes, int n_in,
                              void* d_out, int out_size, void* d_ws, size_t ws_size,
                              hipStream_t stream) {
    const float* x  = (const float*)d_in[0];
    const float* Wt = (const float*)d_in[1];
    const float* Wx = (const float*)d_in[2];
    const float* bh = (const float*)d_in[3];
    const float* Wa = (const float*)d_in[4];
    const float* ba = (const float*)d_in[5];
    float* out = (float*)d_out;

    char* ws = (char*)d_ws;
    float*  q  = (float*)(ws);
    float*  kk = (float*)(ws + 524288);
    __bf16* Xt = (__bf16*)(ws + 1048576);
    __bf16* P  = (__bf16*)(ws + 3145728);

    // 1) q,k GEMVs: B*L=4096 rows, 4 waves/block
    qk_kernel<<<1024, 128, 0, stream>>>(x, Wt, Wx, q, kk);
    // 2) transpose + bf16 convert X
    xt_kernel<<<(BB * LL * DD) / 256, 256, 0, stream>>>(x, Xt);
    // 3) scores + softmax, one block per (b,i) row
    score_softmax_kernel<<<BB * LL, 256, 0, stream>>>(q, kk, bh, Wa, ba, P);
    // 4) V = P @ X with WMMA bf16: 1024 waves (16x64 tiles), 4 waves/block
    pv_wmma_kernel<<<256, 128, 0, stream>>>(P, Xt, out);
}
